// MultiHeadGridAttention2d_3667902070983
// MI455X (gfx1250) — matrix operand order repaired and checked by reading, not compiled, not run on hardware
//
#include <hip/hip_runtime.h>

#define C_ 512
#define NH_ 8
#define KD_ 32
#define P_ 10000
#define B_ 8
#define SCALE_ 0.17677669529663688f   // 32^-0.5

typedef unsigned short ushort_t;
typedef __attribute__((ext_vector_type(16))) __bf16 bf16x16;
typedef __attribute__((ext_vector_type(8)))  float  f32x8;
typedef __attribute__((ext_vector_type(4)))  unsigned int u32x4;
typedef __attribute__((ext_vector_type(8)))  unsigned int u32x8;

__device__ __forceinline__ ushort_t f2bf(float f) {
  unsigned int u = __builtin_bit_cast(unsigned int, f);
  u += 0x7FFFu + ((u >> 16) & 1u);            // round-to-nearest-even
  return (ushort_t)(u >> 16);
}

// ---------------------------------------------------------------- pack f32->bf16 (weights)
__global__ void k_f32_to_bf16(const float* __restrict__ in, ushort_t* __restrict__ out, int n) {
  int i = blockIdx.x * 256 + threadIdx.x;
  if (i < n) out[i] = f2bf(in[i]);
}

// ---------------------------------------------------------------- pack x -> K-pair-packed bf16 dwords
// XP[(b*256 + c/2)*P + p] = { bf16(x[b][c][p]), bf16(x[b][c+1][p]) }
__global__ void k_pack_x_pairs(const float* __restrict__ in, unsigned* __restrict__ out, int n) {
  int idx = blockIdx.x * 256 + threadIdx.x;
  if (idx >= n) return;
  int p  = idx % P_;
  int c2 = (idx / P_) & 255;
  int b  = idx / (P_ * 256);
  const float* src = in + ((size_t)b * C_ + 2 * c2) * P_ + p;
  unsigned lo = f2bf(src[0]);
  unsigned hi = f2bf(src[P_]);
  out[idx] = lo | (hi << 16);
}

// ---------------------------------------------------------------- WMMA GEMM
// Y[b][oc][p] = sc[oc] * sum_c W[oc][c] * X[b][c][p] + bi[oc]
// W: 512x512 bf16 row-major; XP: [B][256][P] K-pair dwords; Y: [B][512][P] f32
// Block 256 thr (8 waves). Tile 128(M) x 256(N), K-step 32.
// Waves: 2(M) x 4(N); each wave owns 64x64 -> 4x4 subtiles of 16x16 (16 WMMA / K-step).
__global__ __launch_bounds__(256) void k_gemm_bf16(
    const ushort_t* __restrict__ W, const unsigned* __restrict__ XP,
    const float* __restrict__ sc, const float* __restrict__ bi,
    float* __restrict__ Y)
{
  __shared__ __align__(16) ushort_t     As[128 * 40];  // row = 32 bf16, padded to 40  (10.0 KB)
  __shared__ __align__(16) unsigned int Bs[256 * 20];  // col = 16 K-pair dwords, padded to 20 (20.0 KB)

  const int tid = threadIdx.x;
  const int lane = tid & 31;
  const int wid  = tid >> 5;
  const int waveM = wid & 1, waveN = wid >> 1;
  const int half = lane >> 4, ln = lane & 15;

  const int p0  = blockIdx.x * 256;
  const int oc0 = blockIdx.y * 128;
  const int b   = blockIdx.z;

  const unsigned* Xb = XP + (size_t)b * 256 * P_;
  float*          Yb = Y + (size_t)b * C_ * P_;

  f32x8 acc[4][4] = {};

  const int am = tid >> 1, ah = tid & 1;           // A loader: row, half-row
  const int bp = min(p0 + tid, P_ - 1);            // B loader: clamped column (no divergence;
                                                   // out-of-range columns discarded in epilogue)

  // per-thread constant addresses for the async A copy
  const unsigned a_lds = (unsigned)(size_t)((char*)As + am * 80 + ah * 32);
  unsigned long long a_gbl = (unsigned long long)(size_t)(W + (size_t)(oc0 + am) * C_ + ah * 16);

  for (int kt = 0; kt < C_; kt += 32) {
    __syncthreads();
    { // A tile: 128 rows x 32 K, async global->LDS (2 x B128 per thread), ASYNCcnt-tracked
      asm volatile("global_load_async_to_lds_b128 %0, %1, off\n\t"
                   "global_load_async_to_lds_b128 %0, %1, off offset:16"
                   :: "v"(a_lds), "v"(a_gbl + (unsigned long long)(2 * kt)) : "memory");
    }
    { // B tile: 16 K-pair rows x 256 N, pre-packed dwords, transposed into padded columns
      const unsigned* xc = Xb + (size_t)(kt >> 1) * P_ + bp;
      #pragma unroll
      for (int kp = 0; kp < 16; kp++)
        Bs[tid * 20 + kp] = xc[(size_t)kp * P_];
    }
    asm volatile("s_wait_asynccnt 0x0" ::: "memory");
    __syncthreads();

    bf16x16 af[4], bfr[4];
    #pragma unroll
    for (int mi = 0; mi < 4; mi++) {       // A frag: lane = M row; half-wave K interleave
      int mrow = waveM * 64 + mi * 16 + ln;
      const char* rp = (const char*)As + mrow * 80 + half * 16;
      u32x4 lo = *(const u32x4*)rp;        // K 0-7  (half0) / 8-15  (half1)
      u32x4 hi = *(const u32x4*)(rp + 32); // K 16-23(half0) / 24-31 (half1)
      u32x8 t;
      #pragma unroll
      for (int q = 0; q < 4; q++) { t[q] = lo[q]; t[q + 4] = hi[q]; }
      af[mi] = __builtin_bit_cast(bf16x16, t);
    }
    #pragma unroll
    for (int ni = 0; ni < 4; ni++) {       // B frag: lane = N col; half-wave K interleave
      int ncol = waveN * 64 + ni * 16 + ln;
      const char* cp = (const char*)Bs + ncol * 80 + half * 16;
      u32x4 lo = *(const u32x4*)cp;
      u32x4 hi = *(const u32x4*)(cp + 32);
      u32x8 t;
      #pragma unroll
      for (int q = 0; q < 4; q++) { t[q] = lo[q]; t[q + 4] = hi[q]; }
      bfr[ni] = __builtin_bit_cast(bf16x16, t);
    }
    #pragma unroll
    for (int mi = 0; mi < 4; mi++)
      #pragma unroll
      for (int ni = 0; ni < 4; ni++)
        acc[mi][ni] = __builtin_amdgcn_wmma_f32_16x16x32_bf16(
            false, af[mi], false, bfr[ni], (short)0, acc[mi][ni], false, false);
  }

  // epilogue: C/D layout -> M = d + 8*half, N = ln
  #pragma unroll
  for (int mi = 0; mi < 4; mi++) {
    #pragma unroll
    for (int ni = 0; ni < 4; ni++) {
      int p = p0 + waveN * 64 + ni * 16 + ln;
      if (p < P_) {
        #pragma unroll
        for (int d = 0; d < 8; d++) {
          int oc = oc0 + waveM * 64 + mi * 16 + d + 8 * half;
          Yb[(size_t)oc * P_ + p] = acc[mi][ni][d] * sc[oc] + bi[oc];
        }
      }
    }
  }
}

// ---------------------------------------------------------------- head-sum of k
// KS[m][b][d][p] = sum_h YALL[m][b][h*64+32+d][p]
__global__ void k_ksum(const float* __restrict__ YALL, float* __restrict__ KS) {
  int by = blockIdx.y;                 // (m*8+b)*32 + d
  int p = blockIdx.x * 256 + threadIdx.x;
  if (p >= P_) return;
  int d = by & 31;
  int mb = by >> 5;
  const float* src = YALL + ((size_t)mb * C_ + 32 + d) * P_ + p;
  float s = 0.f;
  #pragma unroll
  for (int h = 0; h < NH_; h++) s += src[(size_t)h * 64 * P_];
  KS[(size_t)by * P_ + p] = s;
}

// ---------------------------------------------------------------- attention stages 1-2
// Block = (b, H, k*10+l). bufA: v -> stage2 q/ks. bufB: stage1 q/ks -> t1.
__global__ __launch_bounds__(256) void k_attn12(
    const float* __restrict__ YALL, const float* __restrict__ KS, float* __restrict__ T2)
{
  __shared__ float bufA[6400];
  __shared__ float bufB[6400];
  __shared__ float lg[1000];
  __shared__ float aa[1000];

  const int tid = threadIdx.x;
  const int kl = blockIdx.x, H = blockIdx.y, bb = blockIdx.z;

  const float* q1  = YALL + ((size_t)(0 * B_ + bb) * C_ + H * 64) * P_;
  const float* q2  = YALL + ((size_t)(1 * B_ + bb) * C_ + H * 64) * P_;
  const float* v0  = YALL + ((size_t)(4 * B_ + bb) * C_ + H * 64) * P_;
  const float* ks1 = KS + (size_t)(0 * B_ + bb) * 32 * P_;
  const float* ks2 = KS + (size_t)(1 * B_ + bb) * 32 * P_;
  float* t2o = T2 + (size_t)(bb * NH_ + H) * 64 * P_;

  for (int idx = tid; idx < 6400; idx += 256) {          // vt[d][i*10+j]
    int d = idx / 100, ij = idx % 100;
    bufA[idx] = v0[(size_t)d * P_ + (ij / 10) * 1000 + (ij % 10) * 100 + kl];
  }
  for (int idx = tid; idx < 3200; idx += 256) {          // q1[d][i][j], ksum1[d][I][j]
    int d = idx / 100, ij = idx % 100;
    size_t off = (size_t)d * P_ + (ij / 10) * 1000 + (ij % 10) * 100 + kl;
    bufB[idx] = q1[off];
    bufB[3200 + idx] = ks1[off];
  }
  __syncthreads();
  for (int idx = tid; idx < 1000; idx += 256) {          // lg[j*100+I*10+i]
    int j = idx / 100, I = (idx / 10) % 10, i = idx % 10;
    float s = 0.f;
    for (int d = 0; d < 32; d++)
      s += bufB[d * 100 + i * 10 + j] * bufB[3200 + d * 100 + I * 10 + j];
    lg[idx] = s * SCALE_;
  }
  __syncthreads();
  if (tid < 100) {                                       // softmax over i per (j,I)
    int base = tid * 10;
    float m = lg[base];
    for (int i = 1; i < 10; i++) m = fmaxf(m, lg[base + i]);
    float e[10], s = 0.f;
    for (int i = 0; i < 10; i++) { e[i] = __expf(lg[base + i] - m); s += e[i]; }
    float inv = 1.f / s;
    for (int i = 0; i < 10; i++) aa[base + i] = e[i] * inv;
  }
  __syncthreads();
  for (int idx = tid; idx < 6400; idx += 256) {          // t1[d][I][j] -> bufB
    int d = idx / 100, I = (idx % 100) / 10, j = idx % 10;
    float s = 0.f;
    for (int i = 0; i < 10; i++)
      s += bufA[d * 100 + i * 10 + j] * aa[j * 100 + I * 10 + i];
    bufB[d * 100 + I * 10 + j] = s;
  }
  __syncthreads();
  for (int idx = tid; idx < 3200; idx += 256) {          // q2[d][I][j], ksum2[d][I][J] -> bufA
    int d = idx / 100, ij = idx % 100;
    size_t off = (size_t)d * P_ + (ij / 10) * 1000 + (ij % 10) * 100 + kl;
    bufA[idx] = q2[off];
    bufA[3200 + idx] = ks2[off];
  }
  __syncthreads();
  for (int idx = tid; idx < 1000; idx += 256) {          // lg[I*100+J*10+j]
    int I = idx / 100, J = (idx / 10) % 10, j = idx % 10;
    float s = 0.f;
    for (int d = 0; d < 32; d++)
      s += bufA[d * 100 + I * 10 + j] * bufA[3200 + d * 100 + I * 10 + J];
    lg[idx] = s * SCALE_;
  }
  __syncthreads();
  if (tid < 100) {                                       // softmax over j per (I,J)
    int base = tid * 10;
    float m = lg[base];
    for (int j = 1; j < 10; j++) m = fmaxf(m, lg[base + j]);
    float e[10], s = 0.f;
    for (int j = 0; j < 10; j++) { e[j] = __expf(lg[base + j] - m); s += e[j]; }
    float inv = 1.f / s;
    for (int j = 0; j < 10; j++) aa[base + j] = e[j] * inv;
  }
  __syncthreads();
  for (int idx = tid; idx < 6400; idx += 256) {          // t2[d][I][J] -> global
    int d = idx / 100, I = (idx % 100) / 10, J = idx % 10;
    float s = 0.f;
    for (int j = 0; j < 10; j++)
      s += bufB[d * 100 + I * 10 + j] * aa[I * 100 + J * 10 + j];
    t2o[(size_t)d * P_ + I * 1000 + J * 100 + kl] = s;
  }
}

// ---------------------------------------------------------------- attention stages 3-4 + pe + packed-bf16 store
// Block = (b, H, r=I*10+J). bufA: t2 -> stage4 q/ks. bufB: stage3 q/ks -> t3.
__global__ __launch_bounds__(256) void k_attn34(
    const float* __restrict__ YALL, const float* __restrict__ KS,
    const float* __restrict__ T2,
    const float* __restrict__ pew, const float* __restrict__ pes, const float* __restrict__ peb,
    ushort_t* __restrict__ OB)   // K-pair-packed: u16 index = 2*((b*256 + ch/2)*P + pix) + (ch&1)
{
  __shared__ float bufA[6400];
  __shared__ float bufB[6400];
  __shared__ float lg[1000];
  __shared__ float aa[1000];

  const int tid = threadIdx.x;
  const int r = blockIdx.x, H = blockIdx.y, bb = blockIdx.z;
  const int pb = r * 100;

  const float* q3  = YALL + ((size_t)(2 * B_ + bb) * C_ + H * 64) * P_;
  const float* q4  = YALL + ((size_t)(3 * B_ + bb) * C_ + H * 64) * P_;
  const float* v0f = YALL + (size_t)(4 * B_ + bb) * C_ * P_;
  const float* ks3 = KS + (size_t)(2 * B_ + bb) * 32 * P_;
  const float* ks4 = KS + (size_t)(3 * B_ + bb) * 32 * P_;
  const float* t2i = T2 + (size_t)(bb * NH_ + H) * 64 * P_;

  for (int idx = tid; idx < 6400; idx += 256)            // tt[d][k*10+l]
    bufA[idx] = t2i[(size_t)(idx / 100) * P_ + pb + (idx % 100)];
  for (int idx = tid; idx < 3200; idx += 256) {          // q3[d][k][l], ksum3[d][K][l]
    size_t off = (size_t)(idx / 100) * P_ + pb + (idx % 100);
    bufB[idx] = q3[off];
    bufB[3200 + idx] = ks3[off];
  }
  __syncthreads();
  for (int idx = tid; idx < 1000; idx += 256) {          // lg[l*100+K*10+k]
    int l = idx / 100, K = (idx / 10) % 10, k = idx % 10;
    float s = 0.f;
    for (int d = 0; d < 32; d++)
      s += bufB[d * 100 + k * 10 + l] * bufB[3200 + d * 100 + K * 10 + l];
    lg[idx] = s * SCALE_;
  }
  __syncthreads();
  if (tid < 100) {                                       // softmax over k per (l,K)
    int base = tid * 10;
    float m = lg[base];
    for (int k = 1; k < 10; k++) m = fmaxf(m, lg[base + k]);
    float e[10], s = 0.f;
    for (int k = 0; k < 10; k++) { e[k] = __expf(lg[base + k] - m); s += e[k]; }
    float inv = 1.f / s;
    for (int k = 0; k < 10; k++) aa[base + k] = e[k] * inv;
  }
  __syncthreads();
  for (int idx = tid; idx < 6400; idx += 256) {          // t3[d][K][l] -> bufB
    int d = idx / 100, K = (idx % 100) / 10, l = idx % 10;
    float s = 0.f;
    for (int k = 0; k < 10; k++)
      s += bufA[d * 100 + k * 10 + l] * aa[l * 100 + K * 10 + k];
    bufB[d * 100 + K * 10 + l] = s;
  }
  __syncthreads();
  for (int idx = tid; idx < 3200; idx += 256) {          // q4[d][K][l], ksum4[d][K][L] -> bufA
    size_t off = (size_t)(idx / 100) * P_ + pb + (idx % 100);
    bufA[idx] = q4[off];
    bufA[3200 + idx] = ks4[off];
  }
  __syncthreads();
  for (int idx = tid; idx < 1000; idx += 256) {          // lg[K*100+L*10+l]
    int K = idx / 100, L = (idx / 10) % 10, l = idx % 10;
    float s = 0.f;
    for (int d = 0; d < 32; d++)
      s += bufA[d * 100 + K * 10 + l] * bufA[3200 + d * 100 + K * 10 + L];
    lg[idx] = s * SCALE_;
  }
  __syncthreads();
  if (tid < 100) {                                       // softmax over l per (K,L)
    int base = tid * 10;
    float m = lg[base];
    for (int l = 1; l < 10; l++) m = fmaxf(m, lg[base + l]);
    float e[10], s = 0.f;
    for (int l = 0; l < 10; l++) { e[l] = __expf(lg[base + l] - m); s += e[l]; }
    float inv = 1.f / s;
    for (int l = 0; l < 10; l++) aa[base + l] = e[l] * inv;
  }
  __syncthreads();
  for (int idx = tid; idx < 6400; idx += 256) {          // t4 + pe -> OB (packed bf16)
    int d = idx / 100, K = (idx % 100) / 10, L = idx % 10;
    float t4 = 0.f;
    for (int l = 0; l < 10; l++)
      t4 += bufB[d * 100 + K * 10 + l] * aa[K * 100 + L * 10 + l];
    int ch = H * 64 + d;
    int cc = K * 10 + L;
    float pe = 0.f;
    #pragma unroll
    for (int dy = 0; dy < 3; dy++) {
      int ry = r + dy - 1;
      if (ry < 0 || ry >= 100) continue;
      #pragma unroll
      for (int dx = 0; dx < 3; dx++) {
        int cx = cc + dx - 1;
        if (cx < 0 || cx >= 100) continue;
        pe += v0f[(size_t)ch * P_ + ry * 100 + cx] * pew[ch * 9 + dy * 3 + dx];
      }
    }
    float o = t4 + pe * pes[ch] + peb[ch];
    size_t dw = (size_t)(bb * 256 + (ch >> 1)) * P_ + pb + cc;
    OB[dw * 2 + (ch & 1)] = f2bf(o);
  }
}

// ---------------------------------------------------------------- host launcher
extern "C" void kernel_launch(void* const* d_in, const int* in_sizes, int n_in,
                              void* d_out, int out_size, void* d_ws, size_t ws_size,
                              hipStream_t stream) {
  (void)in_sizes; (void)n_in; (void)out_size; (void)ws_size;

  const float* x      = (const float*)d_in[0];
  const float* wsrc6[6] = {(const float*)d_in[1], (const float*)d_in[4], (const float*)d_in[7],
                           (const float*)d_in[10], (const float*)d_in[13], (const float*)d_in[19]};
  const float* ssrc5[5] = {(const float*)d_in[2], (const float*)d_in[5], (const float*)d_in[8],
                           (const float*)d_in[11], (const float*)d_in[14]};
  const float* bsrc5[5] = {(const float*)d_in[3], (const float*)d_in[6], (const float*)d_in[9],
                           (const float*)d_in[12], (const float*)d_in[15]};
  const float* pe_w = (const float*)d_in[16];
  const float* pe_s = (const float*)d_in[17];
  const float* pe_b = (const float*)d_in[18];
  const float* proj_s = (const float*)d_in[20];
  const float* proj_b = (const float*)d_in[21];
  float* out = (float*)d_out;

  char* ws = (char*)d_ws;
  const size_t szXP = (size_t)B_ * 256 * P_ * 4;       //  81,920,000 B (pair-packed dwords)
  const size_t szOB = szXP;                            //  81,920,000 B
  const size_t szW  = (size_t)6 * C_ * C_ * 2;         //   3,145,728 B
  const size_t szY  = (size_t)5 * B_ * C_ * P_ * 4;    // 819,200,000 B
  const size_t szKS = (size_t)4 * B_ * 32 * P_ * 4;    //  40,960,000 B
  unsigned* XP  = (unsigned*)(ws);
  ushort_t* OB  = (ushort_t*)(ws + szXP);
  ushort_t* WPK = (ushort_t*)(ws + szXP + szOB);
  float*    YALL= (float*)(ws + szXP + szOB + szW);
  float*    KS  = (float*)(ws + szXP + szOB + szW + szY);
  float*    T2  = (float*)(ws + szXP + szOB + szW + szY + szKS);

  { int n = B_ * 256 * P_;
    k_pack_x_pairs<<<(n + 255) / 256, 256, 0, stream>>>(x, XP, n); }
  for (int m = 0; m < 6; m++) {
    int n = C_ * C_;
    k_f32_to_bf16<<<(n + 255) / 256, 256, 0, stream>>>(wsrc6[m], WPK + (size_t)m * C_ * C_, n);
  }

  dim3 ggrid((P_ + 255) / 256, 4, B_);                 // 40 x 4 x 8
  for (int m = 0; m < 5; m++)
    k_gemm_bf16<<<ggrid, 256, 0, stream>>>(WPK + (size_t)m * C_ * C_, XP, ssrc5[m], bsrc5[m],
                                           YALL + (size_t)m * B_ * C_ * P_);

  k_ksum<<<dim3((P_ + 255) / 256, 4 * B_ * 32), 256, 0, stream>>>(YALL, KS);
  k_attn12<<<dim3(100, NH_, B_), 256, 0, stream>>>(YALL, KS, T2);
  k_attn34<<<dim3(100, NH_, B_), 256, 0, stream>>>(YALL, KS, T2, pe_w, pe_s, pe_b, OB);

  k_gemm_bf16<<<ggrid, 256, 0, stream>>>(WPK + (size_t)5 * C_ * C_, (const unsigned*)OB,
                                         proj_s, proj_b, out);
}
